// Mamba2_55044300865677
// MI455X (gfx1250) — matrix operand order repaired and checked by reading, not compiled, not run on hardware
//
#include <hip/hip_runtime.h>
#include <hip/hip_bf16.h>
#include <math.h>

typedef __bf16 bf16;
typedef __attribute__((ext_vector_type(16))) __bf16 v16bf;
typedef __attribute__((ext_vector_type(8)))  __bf16 v8bf;
typedef __attribute__((ext_vector_type(8)))  float  v8f;

#define ROWS   8192   // b*L
#define DMODEL 1024
#define DINNER 2048
#define DIPROJ 4384
#define NPAD   4480   // DIPROJ padded to 35*128 so the in-proj GEMM is branch-free
#define CONVCH 2304
#define NHEADS 32
#define HEADD  64
#define DSTATE 128
#define NCHUNK 64
#define CHUNKL 64
#define LSEQ   4096

static __device__ __forceinline__ float sigm_(float x) { return 1.f / (1.f + __expf(-x)); }

static __device__ __forceinline__ v8f acc_zero() {
  v8f a;
#pragma unroll
  for (int i = 0; i < 8; ++i) a[i] = 0.0f;
  return a;
}

// A fragment: 16x32 tile of a row-major [M x K] operand (ISA 16-bit A layout).
// lane m = lane&15 ; lanes 0-15 hold K {0..7,16..23}, lanes 16-31 hold K {8..15,24..31}
static __device__ __forceinline__ v16bf load_frag_a(const bf16* tile, int lda, int lane) {
  const int m  = lane & 15;
  const int kh = (lane >> 4) << 3;
  const bf16* r = tile + (size_t)m * lda + kh;
  v8bf lo = *(const v8bf*)(r);
  v8bf hi = *(const v8bf*)(r + 16);
  v16bf f;
#pragma unroll
  for (int i = 0; i < 8; ++i) { f[i] = lo[i]; f[i + 8] = hi[i]; }
  return f;
}

// B fragment: 32x16 tile supplied as B^T rows (row-major [N x K]); lanes 0-15 hold K {0..7,16..23}, lanes 16-31 hold K {8..15,24..31}.
static __device__ __forceinline__ v16bf load_frag_b(const bf16* tileT, int ldk, int lane) {
  const int n  = lane & 15;
  const int kh = (lane >> 4) << 3;
  const bf16* r = tileT + (size_t)n * ldk + kh;
  v8bf lo = *(const v8bf*)(r);
  v8bf hi = *(const v8bf*)(r + 16);
  v16bf f;
#pragma unroll
  for (int i = 0; i < 8; ++i) { f[i] = lo[i]; f[i + 8] = hi[i]; }
  return f;
}

#define WMMA_BF16(a, b, c) \
  __builtin_amdgcn_wmma_f32_16x16x32_bf16(false, (a), false, (b), (short)0, (c), false, false)

// ---------------- conversion / transpose ----------------
__global__ __launch_bounds__(256) void k_cvt(const float* __restrict__ in, bf16* __restrict__ out, int n) {
  int i = blockIdx.x * 256 + threadIdx.x;
  const int stride = gridDim.x * 256;
  for (; i < n; i += stride) out[i] = (bf16)in[i];
}

// Transposes in[R][C] -> out[Cpad][R] (bf16), zero-filling rows C..Cpad-1.
__global__ __launch_bounds__(256) void k_transpose_cvt(const float* __restrict__ in,
                                                       bf16* __restrict__ out,
                                                       int R, int C, int Cpad) {
  __shared__ float tile[32][33];
  const int tx = threadIdx.x & 31, ty = threadIdx.x >> 5;
  const int c0 = blockIdx.x * 32, r0 = blockIdx.y * 32;
#pragma unroll
  for (int i = 0; i < 32; i += 8) {
    int r = r0 + ty + i, c = c0 + tx;
    tile[ty + i][tx] = (r < R && c < C) ? in[(size_t)r * C + c] : 0.f;
  }
  __syncthreads();
#pragma unroll
  for (int i = 0; i < 32; i += 8) {
    int c = c0 + ty + i, r = r0 + tx;
    if (c < Cpad && r < R) out[(size_t)c * R + r] = (bf16)tile[tx][ty + i];
  }
}

// ---------------- K1: in-projection GEMM + split epilogue (branch-free K-loop) ----------------
__global__ __launch_bounds__(256) void k_inproj(const bf16* __restrict__ ub,
                                                const bf16* __restrict__ WinT,
                                                bf16* __restrict__ zbuf,
                                                bf16* __restrict__ xbc,
                                                float* __restrict__ dtraw) {
  const int lane = threadIdx.x & 31;
  const int wave = threadIdx.x >> 5;
  const int m0 = blockIdx.y * 64 + (wave & 3) * 16;
  const int n0 = blockIdx.x * 128 + (wave >> 2) * 64;
  v8f acc[4];
#pragma unroll
  for (int t = 0; t < 4; ++t) acc[t] = acc_zero();
  for (int k0 = 0; k0 < DMODEL; k0 += 32) {
    v16bf a = load_frag_a(ub + (size_t)m0 * DMODEL + k0, DMODEL, lane);
#pragma unroll
    for (int t = 0; t < 4; ++t) {
      v16bf bq = load_frag_b(WinT + (size_t)(n0 + t * 16) * DMODEL + k0, DMODEL, lane);
      acc[t] = WMMA_BF16(a, bq, acc[t]);
    }
  }
  const int hi = (lane >> 4) << 3;
#pragma unroll
  for (int t = 0; t < 4; ++t) {
    int nt = n0 + t * 16 + (lane & 15);
    if (nt >= DIPROJ) continue;   // store-side guard only; WMMA above is unguarded
#pragma unroll
    for (int r = 0; r < 8; ++r) {
      int row = m0 + r + hi;
      float v = acc[t][r];
      if (nt < DINNER) zbuf[(size_t)row * DINNER + nt] = (bf16)v;
      else if (nt < DINNER + CONVCH) xbc[(size_t)row * CONVCH + (nt - DINNER)] = (bf16)v;
      else dtraw[(size_t)row * NHEADS + (nt - DINNER - CONVCH)] = v;
    }
  }
}

// ---------------- K2: depthwise causal conv1d + SiLU + split ----------------
__global__ __launch_bounds__(256) void k_conv(const bf16* __restrict__ xbc,
                                              const float* __restrict__ conv_w,
                                              const float* __restrict__ conv_b,
                                              bf16* __restrict__ xb,
                                              bf16* __restrict__ Bb,
                                              bf16* __restrict__ Cb) {
  const int ch = blockIdx.y * 256 + threadIdx.x;
  const int row = blockIdx.x;
  const int l = row & (LSEQ - 1);
  float acc = conv_b[ch];
#pragma unroll
  for (int j = 0; j < 4; ++j) {
    int ll = l - 3 + j;
    if (ll >= 0) acc += conv_w[ch * 4 + j] * (float)xbc[(size_t)(row - 3 + j) * CONVCH + ch];
  }
  float s = acc * sigm_(acc);
  if (ch < DINNER) xb[(size_t)row * DINNER + ch] = (bf16)s;
  else if (ch < DINNER + DSTATE) Bb[(size_t)row * DSTATE + (ch - DINNER)] = (bf16)s;
  else Cb[(size_t)row * DSTATE + (ch - DINNER - DSTATE)] = (bf16)s;
}

// ---------------- K3: softplus(dt)+bias, per-chunk cumsum of A*dt ----------------
__global__ void k_dt(float* __restrict__ dtio, const float* __restrict__ dt_bias,
                     const float* __restrict__ A_log, float* __restrict__ Acum,
                     float* __restrict__ Atot) {
  const int bc = blockIdx.x;           // b*64 + c
  const int h = threadIdx.x;           // 0..31
  const float bias = dt_bias[h];
  const float A = -__expf(A_log[h]);
  const int rowbase = (bc >> 6) * LSEQ + (bc & 63) * CHUNKL;
  float run = 0.f;
  for (int l = 0; l < CHUNKL; ++l) {
    float x = dtio[(size_t)(rowbase + l) * NHEADS + h] + bias;
    float sp = (x > 20.f) ? x : log1pf(__expf(x));
    dtio[(size_t)(rowbase + l) * NHEADS + h] = sp;
    run += A * sp;
    Acum[((size_t)bc * NHEADS + h) * CHUNKL + l] = run;
  }
  Atot[(size_t)bc * NHEADS + h] = run;
}

// ---------------- K4: per-chunk state build: states[p][n] = (decay*xd)^T @ B ----------------
__global__ __launch_bounds__(256) void k_chunk_states(const bf16* __restrict__ xb,
                                                      const float* __restrict__ dt,
                                                      const bf16* __restrict__ Bb,
                                                      const float* __restrict__ Acum,
                                                      const float* __restrict__ Atot,
                                                      bf16* __restrict__ cstates) {
  const int h = blockIdx.x & 31;
  const int c = (blockIdx.x >> 5) & 63;
  const int b = blockIdx.x >> 11;
  const int bc = b * NCHUNK + c;
  const int rowbase = b * LSEQ + c * CHUNKL;
  __shared__ __align__(16) bf16 xdT[HEADD][72];
  __shared__ __align__(16) bf16 BT[DSTATE][72];
  __shared__ float sdecay[CHUNKL];
  const int t = threadIdx.x;
  const float* ac = Acum + ((size_t)bc * NHEADS + h) * CHUNKL;
  const float atot = Atot[(size_t)bc * NHEADS + h];
  if (t < CHUNKL) sdecay[t] = __expf(atot - ac[t]);
  __syncthreads();
#pragma unroll
  for (int i = 0; i < 16; ++i) {
    int idx = i * 256 + t;
    int l = idx >> 6, p = idx & 63;
    float v = (float)xb[((size_t)(rowbase + l) * NHEADS + h) * HEADD + p]
              * dt[(size_t)(rowbase + l) * NHEADS + h] * sdecay[l];
    xdT[p][l] = (bf16)v;
  }
#pragma unroll
  for (int i = 0; i < 32; ++i) {
    int idx = i * 256 + t;
    int l = idx >> 7, n = idx & 127;
    BT[n][l] = Bb[(size_t)(rowbase + l) * DSTATE + n];
  }
  __syncthreads();
  const int lane = t & 31, wave = t >> 5;
  const int p0 = (wave & 3) * 16;
  const int n0 = (wave >> 2) * 64;
  v8f acc[4];
#pragma unroll
  for (int tt = 0; tt < 4; ++tt) acc[tt] = acc_zero();
  for (int k0 = 0; k0 < CHUNKL; k0 += 32) {
    v16bf a = load_frag_a(&xdT[p0][k0], 72, lane);
#pragma unroll
    for (int tt = 0; tt < 4; ++tt) {
      v16bf bq = load_frag_b(&BT[n0 + tt * 16][k0], 72, lane);
      acc[tt] = WMMA_BF16(a, bq, acc[tt]);
    }
  }
  bf16* out = cstates + ((size_t)bc * NHEADS + h) * (HEADD * DSTATE);
  const int hi = (lane >> 4) << 3;
#pragma unroll
  for (int tt = 0; tt < 4; ++tt) {
    int n = n0 + tt * 16 + (lane & 15);
#pragma unroll
    for (int r = 0; r < 8; ++r)
      out[(size_t)(p0 + r + hi) * DSTATE + n] = (bf16)acc[tt][r];
  }
}

// ---------------- K5: inter-chunk state scan (exact recurrence) ----------------
__global__ __launch_bounds__(256) void k_scan(const bf16* __restrict__ cstates,
                                              const float* __restrict__ Atot,
                                              bf16* __restrict__ pstates) {
  const int slice = blockIdx.x & 7;
  const int bh = blockIdx.x >> 3;
  const int b = bh >> 5, h = bh & 31;
  const int e0 = slice * 1024 + threadIdx.x * 4;
  float acc[4] = {0.f, 0.f, 0.f, 0.f};
  for (int c = 0; c < NCHUNK; ++c) {
    const size_t base = ((size_t)(b * NCHUNK + c) * NHEADS + h) * (HEADD * DSTATE) + e0;
#pragma unroll
    for (int j = 0; j < 4; ++j) pstates[base + j] = (bf16)acc[j];   // state entering chunk c
    const float g = __expf(Atot[(size_t)(b * NCHUNK + c) * NHEADS + h]);
#pragma unroll
    for (int j = 0; j < 4; ++j) acc[j] = acc[j] * g + (float)cstates[base + j];
  }
}

// ---------------- K6: per-chunk output: Y = (mask*exp)(C B^T) @ xd + diag(exp(Acum)) C state^T + D*x ----
__global__ __launch_bounds__(256) void k_chunk_out(const bf16* __restrict__ Cb,
                                                   const bf16* __restrict__ Bb,
                                                   const bf16* __restrict__ xb,
                                                   const float* __restrict__ dt,
                                                   const float* __restrict__ Acum,
                                                   const bf16* __restrict__ pstates,
                                                   const float* __restrict__ Dp,
                                                   float* __restrict__ ybuf) {
  const int h = blockIdx.x & 31;
  const int c = (blockIdx.x >> 5) & 63;
  const int b = blockIdx.x >> 11;
  const int bc = b * NCHUNK + c;
  const int rowbase = b * LSEQ + c * CHUNKL;
  __shared__ __align__(16) bf16 G[CHUNKL][72];
  __shared__ __align__(16) bf16 xdT[HEADD][72];
  __shared__ float sac[CHUNKL];
  const int t = threadIdx.x, lane = t & 31, wave = t >> 5;
  const float* ac = Acum + ((size_t)bc * NHEADS + h) * CHUNKL;
  if (t < CHUNKL) sac[t] = ac[t];
#pragma unroll
  for (int i = 0; i < 16; ++i) {
    int idx = i * 256 + t;
    int l = idx >> 6, p = idx & 63;
    xdT[p][l] = (bf16)((float)xb[((size_t)(rowbase + l) * NHEADS + h) * HEADD + p]
                       * dt[(size_t)(rowbase + l) * NHEADS + h]);
  }
  __syncthreads();
  const int m0 = (wave & 3) * 16;    // l rows
  const int q0 = (wave >> 2) * 32;   // s cols (GEMM1) / p cols (GEMM2,3)
  const int hi = (lane >> 4) << 3;

  // GEMM1: C @ B^T  (both operands K-contiguous in global memory)
  v8f acc[2]; acc[0] = acc_zero(); acc[1] = acc_zero();
  for (int k0 = 0; k0 < DSTATE; k0 += 32) {
    v16bf a = load_frag_a(Cb + (size_t)(rowbase + m0) * DSTATE + k0, DSTATE, lane);
#pragma unroll
    for (int tt = 0; tt < 2; ++tt) {
      v16bf bq = load_frag_b(Bb + (size_t)(rowbase + q0 + tt * 16) * DSTATE + k0, DSTATE, lane);
      acc[tt] = WMMA_BF16(a, bq, acc[tt]);
    }
  }
#pragma unroll
  for (int tt = 0; tt < 2; ++tt) {
    int s = q0 + tt * 16 + (lane & 15);
#pragma unroll
    for (int r = 0; r < 8; ++r) {
      int l = m0 + r + hi;
      float v = (s <= l) ? acc[tt][r] * __expf(sac[l] - sac[s]) : 0.f;
      G[l][s] = (bf16)v;
    }
  }
  __syncthreads();

  // GEMM3: Y_off = C @ state^T, then row-scale by exp(Acum[l])
  v8f acy[2]; acy[0] = acc_zero(); acy[1] = acc_zero();
  const bf16* st = pstates + ((size_t)bc * NHEADS + h) * (HEADD * DSTATE);
  for (int k0 = 0; k0 < DSTATE; k0 += 32) {
    v16bf a = load_frag_a(Cb + (size_t)(rowbase + m0) * DSTATE + k0, DSTATE, lane);
#pragma unroll
    for (int tt = 0; tt < 2; ++tt) {
      v16bf bq = load_frag_b(st + (size_t)(q0 + tt * 16) * DSTATE + k0, DSTATE, lane);
      acy[tt] = WMMA_BF16(a, bq, acy[tt]);
    }
  }
#pragma unroll
  for (int tt = 0; tt < 2; ++tt)
#pragma unroll
    for (int r = 0; r < 8; ++r) acy[tt][r] *= __expf(sac[m0 + r + hi]);

  // GEMM2: += G @ xd
  for (int k0 = 0; k0 < CHUNKL; k0 += 32) {
    v16bf a = load_frag_a(&G[m0][k0], 72, lane);
#pragma unroll
    for (int tt = 0; tt < 2; ++tt) {
      v16bf bq = load_frag_b(&xdT[q0 + tt * 16][k0], 72, lane);
      acy[tt] = WMMA_BF16(a, bq, acy[tt]);
    }
  }

  const float Dh = Dp[h];
#pragma unroll
  for (int tt = 0; tt < 2; ++tt) {
    int p = q0 + tt * 16 + (lane & 15);
#pragma unroll
    for (int r = 0; r < 8; ++r) {
      int l = m0 + r + hi;
      float xv = (float)xb[((size_t)(rowbase + l) * NHEADS + h) * HEADD + p];
      ybuf[(size_t)(rowbase + l) * DINNER + h * HEADD + p] = acy[tt][r] + Dh * xv;
    }
  }
}

// ---------------- K7: gate with SiLU(z) + RMSNorm ----------------
__global__ __launch_bounds__(256) void k_gatenorm(const float* __restrict__ ybuf,
                                                  const bf16* __restrict__ zbuf,
                                                  const float* __restrict__ norm_w,
                                                  bf16* __restrict__ ybf) {
  const int row = blockIdx.x;
  const int t = threadIdx.x;
  __shared__ float red[256];
  float vals[8]; float ss = 0.f;
#pragma unroll
  for (int i = 0; i < 8; ++i) {
    int cix = t + i * 256;
    float y = ybuf[(size_t)row * DINNER + cix];
    float z = (float)zbuf[(size_t)row * DINNER + cix];
    y *= z * sigm_(z);
    vals[i] = y; ss += y * y;
  }
  red[t] = ss;
  __syncthreads();
  for (int s = 128; s > 0; s >>= 1) {
    if (t < s) red[t] += red[t + s];
    __syncthreads();
  }
  const float inv = rsqrtf(red[0] / (float)DINNER + 1e-5f);
#pragma unroll
  for (int i = 0; i < 8; ++i) {
    int cix = t + i * 256;
    ybf[(size_t)row * DINNER + cix] = (bf16)(vals[i] * inv * norm_w[cix]);
  }
}

// ---------------- K8: output GEMM ----------------
__global__ __launch_bounds__(256) void k_outproj(const bf16* __restrict__ ybf,
                                                 const bf16* __restrict__ WoutT,
                                                 float* __restrict__ out) {
  const int lane = threadIdx.x & 31;
  const int wave = threadIdx.x >> 5;
  const int m0 = blockIdx.y * 64 + (wave & 3) * 16;
  const int n0 = blockIdx.x * 128 + (wave >> 2) * 64;
  v8f acc[4];
#pragma unroll
  for (int t = 0; t < 4; ++t) acc[t] = acc_zero();
  for (int k0 = 0; k0 < DINNER; k0 += 32) {
    v16bf a = load_frag_a(ybf + (size_t)m0 * DINNER + k0, DINNER, lane);
#pragma unroll
    for (int t = 0; t < 4; ++t) {
      v16bf bq = load_frag_b(WoutT + (size_t)(n0 + t * 16) * DINNER + k0, DINNER, lane);
      acc[t] = WMMA_BF16(a, bq, acc[t]);
    }
  }
  const int hi = (lane >> 4) << 3;
#pragma unroll
  for (int t = 0; t < 4; ++t) {
    int n = n0 + t * 16 + (lane & 15);
#pragma unroll
    for (int r = 0; r < 8; ++r)
      out[(size_t)(m0 + r + hi) * DMODEL + n] = acc[t][r];
  }
}

extern "C" void kernel_launch(void* const* d_in, const int* in_sizes, int n_in,
                              void* d_out, int out_size, void* d_ws, size_t ws_size,
                              hipStream_t stream) {
  (void)in_sizes; (void)n_in; (void)out_size; (void)ws_size;
  const float* u       = (const float*)d_in[0];
  const float* W_in    = (const float*)d_in[1];
  const float* conv_w  = (const float*)d_in[2];
  const float* conv_b  = (const float*)d_in[3];
  const float* dt_bias = (const float*)d_in[4];
  const float* A_log   = (const float*)d_in[5];
  const float* Dp      = (const float*)d_in[6];
  const float* norm_w  = (const float*)d_in[7];
  const float* W_out   = (const float*)d_in[8];
  float* out = (float*)d_out;

  size_t off = 0;
  auto take = [&](size_t bytes) -> void* {
    void* p = (char*)d_ws + off;
    off += (bytes + 255) & ~(size_t)255;
    return p;
  };
  bf16*  ub      = (bf16*) take((size_t)ROWS * DMODEL * 2);
  bf16*  WinT    = (bf16*) take((size_t)NPAD * DMODEL * 2);     // zero-padded rows DIPROJ..NPAD-1
  bf16*  WoutT   = (bf16*) take((size_t)DMODEL * DINNER * 2);
  bf16*  zbuf    = (bf16*) take((size_t)ROWS * DINNER * 2);
  bf16*  xbc     = (bf16*) take((size_t)ROWS * CONVCH * 2);
  float* dtbuf   = (float*)take((size_t)ROWS * NHEADS * 4);
  bf16*  xb      = (bf16*) take((size_t)ROWS * DINNER * 2);
  bf16*  Bb      = (bf16*) take((size_t)ROWS * DSTATE * 2);
  bf16*  Cb      = (bf16*) take((size_t)ROWS * DSTATE * 2);
  float* Acum    = (float*)take((size_t)2 * NCHUNK * NHEADS * CHUNKL * 4);
  float* Atot    = (float*)take((size_t)2 * NCHUNK * NHEADS * 4);
  bf16*  cstates = (bf16*) take((size_t)2 * NCHUNK * NHEADS * HEADD * DSTATE * 2);
  bf16*  pstates = (bf16*) take((size_t)2 * NCHUNK * NHEADS * HEADD * DSTATE * 2);
  float* ybuf    = (float*)take((size_t)ROWS * DINNER * 4);
  bf16*  ybf     = (bf16*) take((size_t)ROWS * DINNER * 2);

  // Stage 0: bf16 conversion + weight transposes (W_in^T zero-padded to NPAD rows)
  k_cvt<<<2048, 256, 0, stream>>>(u, ub, ROWS * DMODEL);
  k_transpose_cvt<<<dim3(NPAD / 32, DMODEL / 32), 256, 0, stream>>>(W_in, WinT, DMODEL, DIPROJ, NPAD);
  k_transpose_cvt<<<dim3(DMODEL / 32, DINNER / 32), 256, 0, stream>>>(W_out, WoutT, DINNER, DMODEL, DMODEL);
  // Stage 1: in-projection
  k_inproj<<<dim3(NPAD / 128, ROWS / 64), 256, 0, stream>>>(ub, WinT, zbuf, xbc, dtbuf);
  // Stage 2: conv + SiLU + split
  k_conv<<<dim3(ROWS, CONVCH / 256), 256, 0, stream>>>(xbc, conv_w, conv_b, xb, Bb, Cb);
  // Stage 3: dt softplus + A*dt cumsums
  k_dt<<<2 * NCHUNK, NHEADS, 0, stream>>>(dtbuf, dt_bias, A_log, Acum, Atot);
  // Stage 4: per-chunk states
  k_chunk_states<<<2 * NCHUNK * NHEADS, 256, 0, stream>>>(xb, dtbuf, Bb, Acum, Atot, cstates);
  // Stage 5: inter-chunk scan
  k_scan<<<2 * NHEADS * 8, 256, 0, stream>>>(cstates, Atot, pstates);
  // Stage 6: per-chunk output
  k_chunk_out<<<2 * NCHUNK * NHEADS, 256, 0, stream>>>(Cb, Bb, xb, dtbuf, Acum, pstates, Dp, ybuf);
  // Stage 7: gate + RMSNorm
  k_gatenorm<<<ROWS, 256, 0, stream>>>(ybuf, zbuf, norm_w, ybf);
  // Stage 8: out-projection
  k_outproj<<<dim3(DMODEL / 128, ROWS / 64), 256, 0, stream>>>(ybf, WoutT, out);
}
